// BiMambaEncoder_74474732913280
// MI455X (gfx1250) — compile-verified, matrix operand order checked by reading, not run on hardware
//
#include <hip/hip_runtime.h>
#include <hip/hip_bf16.h>

// ---------------------------------------------------------------------------
// BiMamba encoder for MI455X (gfx1250, wave32, WMMA).
// All six GEMMs: v_wmma_f32_16x16x32_bf16, 128x64 block tile, 32x32 per wave,
// double-buffered LDS staged via GLOBAL_LOAD_ASYNC_TO_LDS_B128 (ASYNCcnt).
// Scan / norms / conv stay fp32 VALU.
// ---------------------------------------------------------------------------

#define B_   2
#define L_   1024
#define DM_  512
#define DI_  1024
#define N_   16
#define KC_  12
#define DTR_ 32

typedef __attribute__((ext_vector_type(16))) __bf16 v16bf;
typedef __attribute__((ext_vector_type(8)))  float  v8f;

union BF16Frag { v16bf v; unsigned short u[16]; };

#if defined(__HIP_DEVICE_COMPILE__) && __has_builtin(__builtin_amdgcn_global_load_async_to_lds_b128)
#define ASYNC_CP 1
typedef int v4i_ __attribute__((ext_vector_type(4)));
typedef __attribute__((address_space(1))) v4i_* as1_v4i;   // global int4*
typedef __attribute__((address_space(3))) v4i_* as3_v4i;   // LDS int4*
#endif

// 16-byte global -> LDS copy; async direct-to-LDS path when available.
__device__ __forceinline__ void cp16_g2l(const void* g, void* l) {
#if defined(ASYNC_CP)
    __builtin_amdgcn_global_load_async_to_lds_b128(
        (as1_v4i)(uintptr_t)g,
        (as3_v4i)(uintptr_t)(unsigned)(uintptr_t)l,
        0, 0);
#else
    *(uint4*)l = *(const uint4*)g;
#endif
}

__device__ __forceinline__ void wait_async_cp() {
#if defined(ASYNC_CP)
    asm volatile("s_wait_asynccnt 0x0" ::: "memory");
#endif
}

__device__ __forceinline__ unsigned short f2bf(float f) {
    unsigned int u = __float_as_uint(f);
    unsigned int r = u + 0x7FFFu + ((u >> 16) & 1u);   // round-to-nearest-even
    if ((u & 0x7F800000u) == 0x7F800000u) r = u;       // inf/nan passthrough
    return (unsigned short)(r >> 16);
}

__device__ __forceinline__ float sigm(float x) { return 1.0f / (1.0f + __expf(-x)); }
__device__ __forceinline__ float softplus_f(float x) { return x > 20.0f ? x : log1pf(__expf(x)); }
__device__ __forceinline__ float selu_f(float x) {
    const float scale = 1.0507009873554805f, alpha = 1.6732632423543772f;
    return x > 0.0f ? scale * x : scale * alpha * (__expf(x) - 1.0f);
}

// ---------------------------------------------------------------------------
// Generic GEMM: out[M,N] = A[M,K] * W[N,K]^T   (A,W bf16, acc f32)
// 256 threads = 8 waves; block tile 128x64; wave tile 32x32 (2x2 wmma);
// K-step 32; double-buffered LDS with async global->LDS prefetch.
// Requires M%128==0, N%64==0, K%32==0 (all shapes here satisfy this).
// ---------------------------------------------------------------------------
constexpr int EPI_F32      = 0;  // outF = v
constexpr int EPI_SOFTPLUS = 1;  // outF = softplus(v + bias[col])
constexpr int EPI_SELU_BF  = 2;  // outB = bf16(selu(v + bias[col]))
constexpr int EPI_BIAS_RES = 3;  // outF = v + bias[col] + resid[row,col]

template <int EPI>
__global__ void __launch_bounds__(256)
gemm_bf16_wmma(const unsigned short* __restrict__ A,
               const unsigned short* __restrict__ W,
               const float* __restrict__ bias,
               const float* __restrict__ resid,
               float* __restrict__ outF,
               unsigned short* __restrict__ outB,
               int M, int N, int Kt)
{
    constexpr int BM = 128, BN = 64, BK = 32, LDT = BK + 8;
    __shared__ unsigned short As[2][BM * LDT];
    __shared__ unsigned short Bs[2][BN * LDT];

    const int tid  = threadIdx.x;
    const int lane = tid & 31;
    const int wv   = tid >> 5;
    const int wm   = wv & 3;          // 4 M-groups of 32 rows
    const int wn   = wv >> 2;         // 2 N-groups of 32 cols
    const int half = lane >> 4;       // K-half selector (ISA 16-bit layout)
    const int r16  = lane & 15;       // row (A) / col (B) within 16x16 tile

    const int bm = blockIdx.x * BM;
    const int bn = blockIdx.y * BN;

    v8f acc[2][2];
#pragma unroll
    for (int mi = 0; mi < 2; ++mi)
#pragma unroll
        for (int ni = 0; ni < 2; ++ni)
            acc[mi][ni] = (v8f){0.f, 0.f, 0.f, 0.f, 0.f, 0.f, 0.f, 0.f};

    // stage one BK-slice into LDS buffer `buf` (async when available)
    auto stage = [&](int buf, int k0) {
        // A tile: 128x32 bf16 = 512 x 16B; 2 per thread
#pragma unroll
        for (int it = 0; it < 2; ++it) {
            int id   = tid + it * 256;
            int row  = id >> 2;
            int colv = (id & 3) << 3;
            cp16_g2l(&A[(size_t)(bm + row) * Kt + k0 + colv],
                     &As[buf][row * LDT + colv]);
        }
        // W tile: 64x32 bf16 = 256 x 16B; 1 per thread
        {
            int row  = tid >> 2;
            int colv = (tid & 3) << 3;
            cp16_g2l(&W[(size_t)(bn + row) * Kt + k0 + colv],
                     &Bs[buf][row * LDT + colv]);
        }
    };

    stage(0, 0);
    wait_async_cp();
    __syncthreads();

    int buf = 0;
    for (int k0 = 0; k0 < Kt; k0 += BK) {
        if (k0 + BK < Kt) stage(buf ^ 1, k0 + BK);   // prefetch next slice

        // fragment gather per ISA 7.12.2 (16-bit 16x32 layout) + 4 wmma
        BF16Frag af[2], bf[2];
#pragma unroll
        for (int mi = 0; mi < 2; ++mi)
#pragma unroll
            for (int e = 0; e < 16; ++e) {
                int k = (e & 7) + ((e >> 3) << 4) + (half << 3);
                af[mi].u[e] = As[buf][(wm * 32 + mi * 16 + r16) * LDT + k];
            }
#pragma unroll
        for (int ni = 0; ni < 2; ++ni)
#pragma unroll
            for (int e = 0; e < 16; ++e) {
                int k = (e & 7) + ((e >> 3) << 4) + (half << 3);
                bf[ni].u[e] = Bs[buf][(wn * 32 + ni * 16 + r16) * LDT + k];
            }
#pragma unroll
        for (int mi = 0; mi < 2; ++mi)
#pragma unroll
            for (int ni = 0; ni < 2; ++ni)
                acc[mi][ni] = __builtin_amdgcn_wmma_f32_16x16x32_bf16(
                    false, af[mi].v, false, bf[ni].v, (short)0,
                    acc[mi][ni], false, false);

        wait_async_cp();     // prefetch landed
        __syncthreads();     // everyone done reading `buf`
        buf ^= 1;
    }

    // epilogue: acc[mi][ni][r] -> row bm+wm*32+mi*16+half*8+r, col bn+wn*32+ni*16+r16
#pragma unroll
    for (int mi = 0; mi < 2; ++mi) {
#pragma unroll
        for (int ni = 0; ni < 2; ++ni) {
            const int rowBase = bm + wm * 32 + mi * 16 + (half << 3);
            const int col     = bn + wn * 32 + ni * 16 + r16;
#pragma unroll
            for (int r = 0; r < 8; ++r) {
                float  v   = acc[mi][ni][r];
                int    row = rowBase + r;
                size_t o   = (size_t)row * N + col;
                if constexpr (EPI == EPI_F32) {
                    outF[o] = v;
                } else if constexpr (EPI == EPI_SOFTPLUS) {
                    outF[o] = softplus_f(v + bias[col]);
                } else if constexpr (EPI == EPI_SELU_BF) {
                    outB[o] = f2bf(selu_f(v + bias[col]));
                } else { // EPI_BIAS_RES
                    outF[o] = v + bias[col] + resid[o];
                }
            }
        }
    }
}

// ---------------------------------------------------------------------------
// Elementwise helpers
// ---------------------------------------------------------------------------
__global__ void f32_to_bf16_k(const float* __restrict__ src,
                              unsigned short* __restrict__ dst, int n) {
    int i = blockIdx.x * blockDim.x + threadIdx.x;
    if (i < n) dst[i] = f2bf(src[i]);
}

// dt_r = dbl[:, :32] -> bf16
__global__ void slice_dtr_k(const float* __restrict__ dbl,
                            unsigned short* __restrict__ dtr, int rows) {
    int i = blockIdx.x * blockDim.x + threadIdx.x;
    if (i < rows * DTR_) {
        int r = i >> 5, c = i & 31;
        dtr[i] = f2bf(dbl[(size_t)r * 64 + c]);
    }
}

// ---------------------------------------------------------------------------
// LayerNorm over DM=512; one block per token row; writes bf16.
// ---------------------------------------------------------------------------
__global__ void __launch_bounds__(256)
ln_bf16_k(const float* __restrict__ x, const float* __restrict__ w,
          const float* __restrict__ b, unsigned short* __restrict__ out) {
    __shared__ float red[256];
    const int row = blockIdx.x, tid = threadIdx.x;
    const float* xr = x + (size_t)row * DM_;
    float v0 = xr[tid], v1 = xr[tid + 256];
    red[tid] = v0 + v1; __syncthreads();
    for (int s = 128; s > 0; s >>= 1) { if (tid < s) red[tid] += red[tid + s]; __syncthreads(); }
    float mean = red[0] * (1.0f / DM_); __syncthreads();
    float d0 = v0 - mean, d1 = v1 - mean;
    red[tid] = d0 * d0 + d1 * d1; __syncthreads();
    for (int s = 128; s > 0; s >>= 1) { if (tid < s) red[tid] += red[tid + s]; __syncthreads(); }
    float rstd = rsqrtf(red[0] * (1.0f / DM_) + 1e-5f);
    out[(size_t)row * DM_ + tid]       = f2bf(d0 * rstd * w[tid]       + b[tid]);
    out[(size_t)row * DM_ + tid + 256] = f2bf(d1 * rstd * w[tid + 256] + b[tid + 256]);
}

// combine fwd + flip(bwd), mask, LayerNorm2 -> bf16
__global__ void __launch_bounds__(256)
combine_ln2_k(const float* __restrict__ outm, const float* __restrict__ mask,
              const float* __restrict__ w, const float* __restrict__ bb,
              unsigned short* __restrict__ out) {
    __shared__ float red[256];
    const int row = blockIdx.x, tid = threadIdx.x;           // row = b*L + l
    const int bb_ = row / L_, l = row % L_;
    const size_t fo = ((size_t)(bb_)       * L_ + l)            * DM_;  // dirb = b
    const size_t bo = ((size_t)(B_ + bb_)  * L_ + (L_ - 1 - l)) * DM_;  // dirb = B+b
    const float m = mask[row];
    float v0 = (outm[fo + tid]       + outm[bo + tid])       * m;
    float v1 = (outm[fo + tid + 256] + outm[bo + tid + 256]) * m;
    red[tid] = v0 + v1; __syncthreads();
    for (int s = 128; s > 0; s >>= 1) { if (tid < s) red[tid] += red[tid + s]; __syncthreads(); }
    float mean = red[0] * (1.0f / DM_); __syncthreads();
    float d0 = v0 - mean, d1 = v1 - mean;
    red[tid] = d0 * d0 + d1 * d1; __syncthreads();
    for (int s = 128; s > 0; s >>= 1) { if (tid < s) red[tid] += red[tid + s]; __syncthreads(); }
    float rstd = rsqrtf(red[0] * (1.0f / DM_) + 1e-5f);
    out[(size_t)row * DM_ + tid]       = f2bf(d0 * rstd * w[tid]       + bb[tid]);
    out[(size_t)row * DM_ + tid + 256] = f2bf(d1 * rstd * w[tid + 256] + bb[tid + 256]);
}

// ---------------------------------------------------------------------------
// Depthwise causal conv (K=12) + bias + SiLU, both directions via index map.
// xz layout: [b][l][2*DI]; output xc: [dirb][l][DI] (dir-local time).
// ---------------------------------------------------------------------------
__global__ void conv_silu_k(const float* __restrict__ xz,
                            const float* __restrict__ cw,
                            const float* __restrict__ cb,
                            float* __restrict__ xc,
                            unsigned short* __restrict__ xcb) {
    const long long i = (long long)blockIdx.x * blockDim.x + threadIdx.x;
    const int d    = (int)(i % DI_);
    const int l    = (int)((i / DI_) % L_);
    const int dirb = (int)(i / ((long long)DI_ * L_));
    const int dir  = dirb / B_, b = dirb % B_;
    float acc = cb[d];
#pragma unroll
    for (int k = 0; k < KC_; ++k) {
        int t = l - (KC_ - 1) + k;
        if (t >= 0) {
            int lm = dir ? (L_ - 1 - t) : t;
            acc += cw[d * KC_ + k] * xz[((size_t)(b * L_ + lm)) * (2 * DI_) + d];
        }
    }
    float s = acc * sigm(acc);
    xc[i]  = s;
    xcb[i] = f2bf(s);
}

// ---------------------------------------------------------------------------
// Selective scan + SiLU(z) gating. One thread per (dirb, d) channel; loop L.
// B/C (32 floats per timestep) staged in LDS once per workgroup.
// ---------------------------------------------------------------------------
__global__ void __launch_bounds__(256)
scan_gate_k(const float* __restrict__ dt, const float* __restrict__ xc,
            const float* __restrict__ dbl, const float* __restrict__ xz,
            const float* __restrict__ A_log, const float* __restrict__ Dv,
            unsigned short* __restrict__ yg) {
    __shared__ float sBC[32];
    const int tid  = threadIdx.x;
    const int d    = blockIdx.x * 256 + tid;
    const int dirb = blockIdx.y;
    const int dir  = dirb / B_, b = dirb % B_;

    float A[N_], h[N_];
#pragma unroll
    for (int n = 0; n < N_; ++n) { A[n] = -__expf(A_log[d * N_ + n]); h[n] = 0.0f; }
    const float Dd = Dv[d];

    for (int l = 0; l < L_; ++l) {
        if (tid < 32)
            sBC[tid] = dbl[((size_t)(dirb * L_ + l)) * 64 + DTR_ + tid];
        __syncthreads();

        const size_t idx = ((size_t)(dirb * L_ + l)) * DI_ + d;
        const float dtv = dt[idx], xv = xc[idx];
        float y = 0.0f;
#pragma unroll
        for (int n = 0; n < N_; ++n) {
            float a = __expf(dtv * A[n]);
            h[n] = a * h[n] + dtv * sBC[n] * xv;
            y += h[n] * sBC[16 + n];
        }
        y += xv * Dd;

        const int lm = dir ? (L_ - 1 - l) : l;
        const float z = xz[((size_t)(b * L_ + lm)) * (2 * DI_) + DI_ + d];
        y *= z * sigm(z);
        yg[idx] = f2bf(y);
        __syncthreads();
    }
}

// ---------------------------------------------------------------------------
// Host-side launch
// ---------------------------------------------------------------------------
extern "C" void kernel_launch(void* const* d_in, const int* in_sizes, int n_in,
                              void* d_out, int out_size, void* d_ws, size_t ws_size,
                              hipStream_t stream) {
    const float* x     = (const float*)d_in[0];
    const float* mask  = (const float*)d_in[1];
    const float* n1w   = (const float*)d_in[2];
    const float* n1b   = (const float*)d_in[3];
    const float* n2w   = (const float*)d_in[4];
    const float* n2b   = (const float*)d_in[5];
    const float* ff1w  = (const float*)d_in[6];
    const float* ff1b  = (const float*)d_in[7];
    const float* ff2w  = (const float*)d_in[8];
    const float* ff2b  = (const float*)d_in[9];
    const float* inpw  = (const float*)d_in[10];
    const float* convw = (const float*)d_in[11];
    const float* convb = (const float*)d_in[12];
    const float* xpw   = (const float*)d_in[13];
    const float* dtpw  = (const float*)d_in[14];
    const float* Alog  = (const float*)d_in[15];
    const float* Dp    = (const float*)d_in[16];
    const float* outpw = (const float*)d_in[17];
    const float* dtpb  = (const float*)d_in[18];
    float* out = (float*)d_out;

    const int M1 = B_ * L_;       // 2048 tokens
    const int M2 = 2 * B_ * L_;   // 4096 dir-batched tokens

    // workspace carve-out (256B-aligned)
    char* ws = (char*)d_ws;
    size_t off = 0;
    auto alloc = [&](size_t bytes) -> void* {
        void* p = ws + off;
        off = (off + bytes + 255) & ~(size_t)255;
        return p;
    };
    unsigned short* xn_bf   = (unsigned short*)alloc((size_t)M1 * DM_ * 2);
    unsigned short* w_in    = (unsigned short*)alloc((size_t)2 * DI_ * DM_ * 2);
    unsigned short* w_xp    = (unsigned short*)alloc((size_t)64 * DI_ * 2);
    unsigned short* w_dtp   = (unsigned short*)alloc((size_t)DI_ * DTR_ * 2);
    unsigned short* w_op    = (unsigned short*)alloc((size_t)DM_ * DI_ * 2);
    unsigned short* w_f1    = (unsigned short*)alloc((size_t)4 * DM_ * DM_ * 2);
    unsigned short* w_f2    = (unsigned short*)alloc((size_t)DM_ * 4 * DM_ * 2);
    float*          xz      = (float*)alloc((size_t)M1 * 2 * DI_ * 4);
    float*          xc      = (float*)alloc((size_t)M2 * DI_ * 4);
    unsigned short* xc_bf   = (unsigned short*)alloc((size_t)M2 * DI_ * 2);
    float*          dbl     = (float*)alloc((size_t)M2 * 64 * 4);
    unsigned short* dtr_bf  = (unsigned short*)alloc((size_t)M2 * DTR_ * 2);
    float*          dt      = (float*)alloc((size_t)M2 * DI_ * 4);
    unsigned short* yg_bf   = (unsigned short*)alloc((size_t)M2 * DI_ * 2);
    float*          outm    = (float*)alloc((size_t)M2 * DM_ * 4);
    unsigned short* mo_bf   = (unsigned short*)alloc((size_t)M1 * DM_ * 2);
    unsigned short* h_bf    = (unsigned short*)alloc((size_t)M1 * 4 * DM_ * 2);
    (void)ws_size; (void)in_sizes; (void)n_in; (void)out_size;

    auto cvt = [&](const float* s, unsigned short* d, int n) {
        f32_to_bf16_k<<<(n + 255) / 256, 256, 0, stream>>>(s, d, n);
    };
    cvt(inpw,  w_in,  2 * DI_ * DM_);
    cvt(xpw,   w_xp,  64 * DI_);
    cvt(dtpw,  w_dtp, DI_ * DTR_);
    cvt(outpw, w_op,  DM_ * DI_);
    cvt(ff1w,  w_f1,  4 * DM_ * DM_);
    cvt(ff2w,  w_f2,  DM_ * 4 * DM_);

    // 1) LayerNorm1 -> bf16
    ln_bf16_k<<<M1, 256, 0, stream>>>(x, n1w, n1b, xn_bf);

    // 2) in_proj: xz[M1, 2048] = xn * in_proj_w^T
    gemm_bf16_wmma<EPI_F32><<<dim3(M1 / 128, (2 * DI_) / 64), 256, 0, stream>>>(
        xn_bf, w_in, nullptr, nullptr, xz, nullptr, M1, 2 * DI_, DM_);

    // 3) depthwise causal conv + SiLU (both directions)
    conv_silu_k<<<(M2 * DI_) / 256, 256, 0, stream>>>(xz, convw, convb, xc, xc_bf);

    // 4) x_proj: dbl[M2, 64] = xc * x_proj_w^T
    gemm_bf16_wmma<EPI_F32><<<dim3(M2 / 128, 64 / 64), 256, 0, stream>>>(
        xc_bf, w_xp, nullptr, nullptr, dbl, nullptr, M2, 64, DI_);

    // 5) dt = softplus(dt_r * dt_proj_w^T + dt_proj_b)
    slice_dtr_k<<<(M2 * DTR_) / 256, 256, 0, stream>>>(dbl, dtr_bf, M2);
    gemm_bf16_wmma<EPI_SOFTPLUS><<<dim3(M2 / 128, DI_ / 64), 256, 0, stream>>>(
        dtr_bf, w_dtp, dtpb, nullptr, dt, nullptr, M2, DI_, DTR_);

    // 6) selective scan + SiLU(z) gating -> bf16
    scan_gate_k<<<dim3(DI_ / 256, 2 * B_), 256, 0, stream>>>(
        dt, xc, dbl, xz, Alog, Dp, yg_bf);

    // 7) out_proj: outm[M2, 512] = yg * out_proj_w^T
    gemm_bf16_wmma<EPI_F32><<<dim3(M2 / 128, DM_ / 64), 256, 0, stream>>>(
        yg_bf, w_op, nullptr, nullptr, outm, nullptr, M2, DM_, DI_);

    // 8) combine fwd + flip(bwd), mask, LayerNorm2 -> bf16
    combine_ln2_k<<<M1, 256, 0, stream>>>(outm, mask, n2w, n2b, mo_bf);

    // 9) FF1: h = selu(mo * ff1_w^T + ff1_b) -> bf16
    gemm_bf16_wmma<EPI_SELU_BF><<<dim3(M1 / 128, (4 * DM_) / 64), 256, 0, stream>>>(
        mo_bf, w_f1, ff1b, nullptr, nullptr, h_bf, M1, 4 * DM_, DM_);

    // 10) FF2 + bias + residual(x) -> d_out (f32)
    gemm_bf16_wmma<EPI_BIAS_RES><<<dim3(M1 / 128, DM_ / 64), 256, 0, stream>>>(
        h_bf, w_f2, ff2b, x, out, nullptr, M1, DM_, 4 * DM_);
}
